// DecoderLayer_42210938585768
// MI455X (gfx1250) — matrix operand order repaired and checked by reading, not compiled, not run on hardware
//
#include <hip/hip_runtime.h>
#include <hip/hip_bf16.h>

// ---------------------------------------------------------------------------
// RNMT+ decoder layer for MI455X (gfx1250, wave32, WMMA bf16 + async LDS copy)
//   B=32, S=256, ISIZE=1024, OSIZE=1024, K_x=2048, K_h=1024, N=4096
// ---------------------------------------------------------------------------

typedef __attribute__((ext_vector_type(16))) __bf16 v16bf;
typedef __attribute__((ext_vector_type(8)))  float  v8f;
typedef __attribute__((ext_vector_type(8)))  unsigned short us8;

#define BB 32
#define SS 256
#define NN 4096
#define KX 2048
#define KH 1024

__device__ __forceinline__ unsigned short f32_to_bf16(float f) {
    unsigned u = __float_as_uint(f);
    u += 0x7FFFu + ((u >> 16) & 1u);          // round-to-nearest-even
    return (unsigned short)(u >> 16);
}

// A fragment (16x32 bf16, wave32): lane = (half<<4)|m ; elements 0..7 hold
// K = 8*half+0..7, elements 8..15 hold K = 16+8*half+0..7  (within the row).
__device__ __forceinline__ v16bf load_a_frag(const unsigned short* __restrict__ row,
                                             int half) {
    union { v16bf v; us8 h[2]; } u;
    u.h[0] = *(const us8*)(row + 8 * half);
    u.h[1] = *(const us8*)(row + 16 + 8 * half);
    return u.v;
}

// B fragment (32x16 bf16, wave32): lane = (half<<4)|n ; column n of the tile,
// elements 0..7 hold K = 8*half+0..7, elements 8..15 hold K = 16+8*half+0..7
// (W stored transposed: Wt[n][k] = W[k][n]).
__device__ __forceinline__ v16bf load_b_frag(const unsigned short* __restrict__ row,
                                             int half) {
    union { v16bf v; us8 h[2]; } u;
    const unsigned short* p = row + 8 * half;
    u.h[0] = *(const us8*)(p);
    u.h[1] = *(const us8*)(p + 16);
    return u.v;
}

__device__ __forceinline__ void async_b128_to_lds(const unsigned short* gptr,
                                                  const unsigned short* lptr) {
    unsigned lds_addr = (unsigned)(unsigned long long)lptr;   // low 32b = LDS addr
    unsigned long long ga = (unsigned long long)gptr;
    asm volatile("global_load_async_to_lds_b128 %0, %1, off"
                 :: "v"(lds_addr), "v"(ga) : "memory");
}

__device__ __forceinline__ void wait_asynccnt0() {
#if __has_builtin(__builtin_amdgcn_s_wait_asynccnt)
    __builtin_amdgcn_s_wait_asynccnt(0);
#else
    asm volatile("s_wait_asynccnt 0x0" ::: "memory");
#endif
}

// ---------------------------------------------------------------------------
// Pre-pass 1: X = concat(inputo, attn) -> bf16, rows r = b*S+s, K = 2048
// ---------------------------------------------------------------------------
__global__ void convert_x_kernel(const float* __restrict__ inputo,
                                 const float* __restrict__ attn,
                                 unsigned short* __restrict__ Xb) {
    const int total = BB * SS * KX;   // 16,777,216
    for (int idx = blockIdx.x * blockDim.x + threadIdx.x; idx < total;
         idx += gridDim.x * blockDim.x) {
        int r = idx >> 11;            // row (b*S+s)
        int k = idx & (KX - 1);
        float v = (k < 1024) ? inputo[(size_t)r * 1024 + k]
                             : attn[(size_t)r * 1024 + (k - 1024)];
        Xb[idx] = f32_to_bf16(v);
    }
}

// ---------------------------------------------------------------------------
// Pre-pass 2: W [3072,4096] f32 -> WtX [4096,2048] bf16, WtH [4096,1024] bf16
// ---------------------------------------------------------------------------
__global__ void transpose_w_kernel(const float* __restrict__ W,
                                   unsigned short* __restrict__ WtX,
                                   unsigned short* __restrict__ WtH) {
    const int total = 3072 * NN;      // 12,582,912
    for (int idx = blockIdx.x * blockDim.x + threadIdx.x; idx < total;
         idx += gridDim.x * blockDim.x) {
        int k = idx >> 12;            // 0..3071
        int n = idx & (NN - 1);
        unsigned short h = f32_to_bf16(W[idx]);
        if (k < KX) WtX[(size_t)n * KX + k] = h;
        else        WtH[(size_t)n * KH + (k - KX)] = h;
    }
}

// ---------------------------------------------------------------------------
// Pre-pass 3: broadcast init_hx/init_cx into hb (bf16) and c (f32)
// ---------------------------------------------------------------------------
__global__ void init_state_kernel(const float* __restrict__ init_hx,
                                  const float* __restrict__ init_cx,
                                  unsigned short* __restrict__ hb,
                                  float* __restrict__ c) {
    int idx = blockIdx.x * blockDim.x + threadIdx.x;   // 32*1024 threads
    if (idx < BB * KH) {
        hb[idx] = f32_to_bf16(init_hx[idx & (KH - 1)]);
        c[idx]  = init_cx[idx & (KH - 1)];
    }
}

// ---------------------------------------------------------------------------
// Phase 1 GEMM: gx = X @ W_x   (M=8192, N=4096, K=2048), bf16 WMMA, f32 acc.
// 256-thread block computes a 128x128 macro-tile. Per k-step of 32, the
// 128x32 A and B tiles are staged into LDS with async global->LDS copies.
// Manual ping-pong (k-loop unrolled by 2, compile-time buffer indices) keeps
// the 64 accumulator VGPRs pinned across iterations: no rotate copies, no
// WMMA->VALU hazard NOPs in the steady state.
// 8 waves laid out 4M x 2N; each wave owns 32M x 64N (8 f32 accumulators).
// Output stored time-major: gx[(s*32 + b)*4096 + col].
// ---------------------------------------------------------------------------
__global__ void __launch_bounds__(256)
gemm_x_kernel(const unsigned short* __restrict__ Xb,
              const unsigned short* __restrict__ WtX,
              float* __restrict__ gx) {
    __shared__ unsigned short As[2][128 * 32];   // 16 KB
    __shared__ unsigned short Bs[2][128 * 32];   // 16 KB

    const int tid  = threadIdx.x;
    const int wv   = tid >> 5;
    const int lane = tid & 31;
    const int wm   = wv & 3;        // 4 m-blocks of 32 rows
    const int wn   = wv >> 2;       // 2 n-blocks of 64 cols
    const int half = lane >> 4;
    const int l16  = lane & 15;
    const int mb   = blockIdx.x & 63;   // 64 m-macros
    const int nb   = blockIdx.x >> 6;   // 32 n-macros
    const int gm0  = mb * 128;
    const int gn0  = nb * 128;

    // Copy plan: 512 16B chunks per 128x32 tile; this thread moves chunks
    // tid and tid+256 of both A and B each k-step.
    const int r0 = tid >> 2,          c0 = (tid & 3) * 8;
    const int r1 = (tid + 256) >> 2,  c1 = ((tid + 256) & 3) * 8;
    const unsigned short* Ag0 = Xb  + (size_t)(gm0 + r0) * KX + c0;
    const unsigned short* Ag1 = Xb  + (size_t)(gm0 + r1) * KX + c1;
    const unsigned short* Bg0 = WtX + (size_t)(gn0 + r0) * KX + c0;
    const unsigned short* Bg1 = WtX + (size_t)(gn0 + r1) * KX + c1;

    v8f acc[2][4];
#pragma unroll
    for (int mt = 0; mt < 2; ++mt)
#pragma unroll
        for (int nn = 0; nn < 4; ++nn) acc[mt][nn] = (v8f){};

#define ISSUE(buf, k0)                                          \
    do {                                                        \
        async_b128_to_lds(Ag0 + (k0), &As[buf][r0 * 32 + c0]);  \
        async_b128_to_lds(Ag1 + (k0), &As[buf][r1 * 32 + c1]);  \
        async_b128_to_lds(Bg0 + (k0), &Bs[buf][r0 * 32 + c0]);  \
        async_b128_to_lds(Bg1 + (k0), &Bs[buf][r1 * 32 + c1]);  \
    } while (0)

#define COMPUTE(buf)                                                          \
    do {                                                                      \
        v16bf a0 = load_a_frag(&As[buf][(wm * 32      + l16) * 32], half);    \
        v16bf a1 = load_a_frag(&As[buf][(wm * 32 + 16 + l16) * 32], half);    \
        _Pragma("unroll")                                                     \
        for (int nn = 0; nn < 4; ++nn) {                                      \
            v16bf b = load_b_frag(&Bs[buf][(wn * 64 + nn * 16 + l16) * 32],   \
                                  half);                                      \
            acc[0][nn] = __builtin_amdgcn_wmma_f32_16x16x32_bf16(             \
                false, a0, false, b, (short)0, acc[0][nn], false, false);     \
            acc[1][nn] = __builtin_amdgcn_wmma_f32_16x16x32_bf16(             \
                false, a1, false, b, (short)0, acc[1][nn], false, false);     \
        }                                                                     \
    } while (0)

    // prologue: fill buffer 0
    ISSUE(0, 0);
    wait_asynccnt0();
    __syncthreads();

    for (int kk = 0; kk < 64; kk += 2) {
        ISSUE(1, (kk + 1) * 32);       // prefetch odd buffer
        COMPUTE(0);
        wait_asynccnt0();              // odd buffer landed (this wave's copies)
        __syncthreads();               // ...and everyone done reading buffer 0

        if (kk + 2 < 64) ISSUE(0, (kk + 2) * 32);   // prefetch even buffer
        COMPUTE(1);
        wait_asynccnt0();
        __syncthreads();
    }
#undef ISSUE
#undef COMPUTE

    // D layout: lane (half, n=l16); VGPR v -> row m = v + 8*half
#pragma unroll
    for (int mt = 0; mt < 2; ++mt) {
#pragma unroll
        for (int v = 0; v < 8; ++v) {
            int r = gm0 + wm * 32 + mt * 16 + v + 8 * half;  // global row b*S+s
            int b = r >> 8;
            int s = r & 255;
            size_t base = ((size_t)(s * BB + b)) * NN + gn0 + wn * 64 + l16;
#pragma unroll
            for (int nn = 0; nn < 4; ++nn)
                gx[base + nn * 16] = acc[mt][nn][v];
        }
    }
}

// ---------------------------------------------------------------------------
// Recurrent step GEMM: g = h @ W_h + gx[t] + bias   (M=32, N=4096, K=1024)
// 64 waves (16 blocks x 4 waves); each wave owns a 32M x 64N strip so every
// WtH row is read exactly once per step (8 MB/step, L2-resident).
// ---------------------------------------------------------------------------
__global__ void __launch_bounds__(128)
step_gemm_kernel(const unsigned short* __restrict__ hb,
                 const unsigned short* __restrict__ WtH,
                 const float* __restrict__ gx,
                 const float* __restrict__ bias,
                 float* __restrict__ g, int t) {
    const int ns   = blockIdx.x * 4 + (threadIdx.x >> 5);  // 0..63 n-strips
    const int lane = threadIdx.x & 31;
    const int half = lane >> 4;
    const int l16  = lane & 15;

    const unsigned short* arow0 = hb + (size_t)(l16) * KH;
    const unsigned short* arow1 = hb + (size_t)(16 + l16) * KH;
    const unsigned short* brow[4];
#pragma unroll
    for (int nn = 0; nn < 4; ++nn)
        brow[nn] = WtH + (size_t)(ns * 64 + nn * 16 + l16) * KH;

    v8f acc[2][4];
#pragma unroll
    for (int mt = 0; mt < 2; ++mt)
#pragma unroll
        for (int nn = 0; nn < 4; ++nn) acc[mt][nn] = (v8f){};

    for (int k0 = 0; k0 < KH; k0 += 32) {
        v16bf a0 = load_a_frag(arow0 + k0, half);
        v16bf a1 = load_a_frag(arow1 + k0, half);
#pragma unroll
        for (int nn = 0; nn < 4; ++nn) {
            v16bf b = load_b_frag(brow[nn] + k0, half);
            acc[0][nn] = __builtin_amdgcn_wmma_f32_16x16x32_bf16(
                false, a0, false, b, (short)0, acc[0][nn], false, false);
            acc[1][nn] = __builtin_amdgcn_wmma_f32_16x16x32_bf16(
                false, a1, false, b, (short)0, acc[1][nn], false, false);
        }
    }

#pragma unroll
    for (int mt = 0; mt < 2; ++mt) {
#pragma unroll
        for (int v = 0; v < 8; ++v) {
            int m = mt * 16 + v + 8 * half;       // batch row 0..31
            size_t coln = (size_t)ns * 64 + l16;
            size_t gxb  = ((size_t)(t * BB + m)) * NN + coln;
            size_t gb   = ((size_t)m) * NN + coln;
#pragma unroll
            for (int nn = 0; nn < 4; ++nn)
                g[gb + nn * 16] = acc[mt][nn][v] + gx[gxb + nn * 16]
                                + bias[coln + nn * 16];
        }
    }
}

// ---------------------------------------------------------------------------
// Recurrent step pointwise: LayerNorm over 4096 gates, LSTM update, residual.
// One workgroup (256 threads) per batch row.
// ---------------------------------------------------------------------------
__device__ __forceinline__ float sigmoidf(float x) { return 1.0f / (1.0f + __expf(-x)); }

__global__ void __launch_bounds__(256)
step_pointwise_kernel(const float* __restrict__ g,
                      const float* __restrict__ ln_g,
                      const float* __restrict__ ln_b,
                      float* __restrict__ c,
                      unsigned short* __restrict__ hb,
                      const float* __restrict__ inputo,
                      float* __restrict__ out, int t) {
    const int r = blockIdx.x;        // batch row
    const int tid = threadIdx.x;
    const float* gr = g + (size_t)r * NN;

    __shared__ float s_sum[256];
    __shared__ float s_sq[256];

    float sum = 0.f, sq = 0.f;
#pragma unroll
    for (int i = tid; i < NN; i += 256) {
        float x = gr[i];
        sum += x; sq += x * x;
    }
    s_sum[tid] = sum; s_sq[tid] = sq;
    __syncthreads();
    for (int off = 128; off > 0; off >>= 1) {
        if (tid < off) { s_sum[tid] += s_sum[tid + off]; s_sq[tid] += s_sq[tid + off]; }
        __syncthreads();
    }
    const float mu = s_sum[0] * (1.0f / NN);
    const float var = s_sq[0] * (1.0f / NN) - mu * mu;
    const float rs = rsqrtf(var + 1e-5f);

    for (int j = tid; j < KH; j += 256) {
        float gi = (gr[        j] - mu) * rs * ln_g[        j] + ln_b[        j];
        float gf = (gr[1024 + j] - mu) * rs * ln_g[1024 + j] + ln_b[1024 + j];
        float go = (gr[2048 + j] - mu) * rs * ln_g[2048 + j] + ln_b[2048 + j];
        float gc = (gr[3072 + j] - mu) * rs * ln_g[3072 + j] + ln_b[3072 + j];
        float iv = sigmoidf(gi);
        float fv = sigmoidf(gf);
        float ov = sigmoidf(go);
        float hd = tanhf(gc);
        size_t ci = (size_t)r * KH + j;
        float cn = fv * c[ci] + iv * hd;
        c[ci] = cn;
        float h = ov * cn;                    // RNMT+ : no tanh on cell
        hb[ci] = f32_to_bf16(h);
        size_t oi = ((size_t)r * SS + t) * KH + j;
        out[oi] = h + inputo[oi];
    }
}

// ---------------------------------------------------------------------------
// Host launcher. Workspace layout (bytes, all 4 KB aligned):
//   [0,32M)    Xb   bf16 [8192,2048]
//   [32M,48M)  WtX  bf16 [4096,2048]
//   [48M,56M)  WtH  bf16 [4096,1024]
//   [56M,184M) gx   f32  [256,32,4096]   (time-major)
//   then g f32 [32,4096], hb bf16 [32,1024], c f32 [32,1024]
// ---------------------------------------------------------------------------
extern "C" void kernel_launch(void* const* d_in, const int* in_sizes, int n_in,
                              void* d_out, int out_size, void* d_ws, size_t ws_size,
                              hipStream_t stream) {
    const float* inputo  = (const float*)d_in[0];
    const float* attn    = (const float*)d_in[1];
    const float* W       = (const float*)d_in[2];
    const float* bias    = (const float*)d_in[3];
    const float* ln_g    = (const float*)d_in[4];
    const float* ln_b    = (const float*)d_in[5];
    const float* init_hx = (const float*)d_in[6];
    const float* init_cx = (const float*)d_in[7];
    float* out = (float*)d_out;

    char* w = (char*)d_ws;
    unsigned short* Xb  = (unsigned short*)(w + 0);
    unsigned short* WtX = (unsigned short*)(w + 33554432ull);
    unsigned short* WtH = (unsigned short*)(w + 50331648ull);
    float*          gx  = (float*)(w + 58720256ull);
    float*          g   = (float*)(w + 192937984ull);
    unsigned short* hb  = (unsigned short*)(w + 193462272ull);
    float*          c   = (float*)(w + 193527808ull);

    convert_x_kernel  <<<4096, 256, 0, stream>>>(inputo, attn, Xb);
    transpose_w_kernel<<<4096, 256, 0, stream>>>(W, WtX, WtH);
    init_state_kernel <<<128, 256, 0, stream>>>(init_hx, init_cx, hb, c);
    gemm_x_kernel     <<<2048, 256, 0, stream>>>(Xb, WtX, gx);

    for (int t = 0; t < SS; ++t) {
        step_gemm_kernel     <<<16, 128, 0, stream>>>(hb, WtH, gx, bias, g, t);
        step_pointwise_kernel<<<32, 256, 0, stream>>>(g, ln_g, ln_b, c, hb, inputo, out, t);
    }
}
